// GCNReg_42460046688816
// MI455X (gfx1250) — matrix operand order repaired and checked by reading, not compiled, not run on hardware
//
#include <hip/hip_runtime.h>
#include <hip/hip_bf16.h>

#define N_NODES   100000
#define N_EDGES   1600000
#define D         128
#define N_GRAPHS  64

typedef __attribute__((ext_vector_type(16))) __bf16 v16bf;
typedef __attribute__((ext_vector_type(8)))  float  v8f;

union BfFrag { uint4 u[2]; v16bf v; };

static __device__ __forceinline__ unsigned short f32_to_bf16(float f) {
    unsigned int u = __float_as_uint(f);
    u += 0x7FFFu + ((u >> 16) & 1u);   // round to nearest even
    return (unsigned short)(u >> 16);
}

// ---------------- degree / norms ----------------
__global__ void deg_kernel(const int* __restrict__ src, const int* __restrict__ dst,
                           float* __restrict__ outdeg, float* __restrict__ indeg) {
    int e = blockIdx.x * blockDim.x + threadIdx.x;
    if (e >= N_EDGES) return;
    atomicAdd(&outdeg[src[e]], 1.0f);
    atomicAdd(&indeg[dst[e]], 1.0f);
}

__global__ void norm_kernel(float* __restrict__ outn, float* __restrict__ inn) {
    int i = blockIdx.x * blockDim.x + threadIdx.x;
    if (i >= N_NODES) return;
    outn[i] = rsqrtf(fmaxf(outn[i], 1.0f));
    inn[i]  = rsqrtf(fmaxf(inn[i], 1.0f));
}

// ---------------- weight transpose + bf16 convert: Wt[n*128+k] = bf16(W[k*128+n]) --------
__global__ void wconv_kernel(const float* __restrict__ W1, const float* __restrict__ W2,
                             unsigned short* __restrict__ Wt1, unsigned short* __restrict__ Wt2) {
    int tid = blockIdx.x * blockDim.x + threadIdx.x;   // 2 * 128*128
    if (tid >= 2 * D * D) return;
    int which = tid >> 14;
    int idx = tid & (D * D - 1);
    int k = idx >> 7, n = idx & 127;
    const float* W = which ? W2 : W1;
    unsigned short* Wt = which ? Wt2 : Wt1;
    Wt[n * D + k] = f32_to_bf16(W[k * D + n]);
}

// ---------------- hn = X * out_norm (float4 per thread) ----------------
__global__ void scale_kernel(const float4* __restrict__ X4, const float* __restrict__ outn,
                             float4* __restrict__ Y4) {
    long long tid = (long long)blockIdx.x * blockDim.x + threadIdx.x;  // N*32
    if (tid >= (long long)N_NODES * 32) return;
    int node = (int)(tid >> 5);
    float s = outn[node];
    float4 v = X4[tid];
    v.x *= s; v.y *= s; v.z *= s; v.w *= s;
    Y4[tid] = v;
}

// ---------------- edge scatter: agg[dst] += hn[src] (wave per edge, float4 per lane) ------
__global__ void scatter_kernel(const float4* __restrict__ hn4, const int* __restrict__ src,
                               const int* __restrict__ dst, float* __restrict__ agg) {
    long long tid = (long long)blockIdx.x * blockDim.x + threadIdx.x;  // E*32
    if (tid >= (long long)N_EDGES * 32) return;
    int e = (int)(tid >> 5);
    int c = (int)(tid & 31);
    int s = src[e], d = dst[e];
    float4 v = hn4[(size_t)s * 32 + c];
    float* ag = agg + (size_t)d * D + c * 4;
    atomicAdd(ag + 0, v.x);
    atomicAdd(ag + 1, v.y);
    atomicAdd(ag + 2, v.z);
    atomicAdd(ag + 3, v.w);
}

// ---------------- abf = bf16(agg * in_norm) ----------------
__global__ void finalize_kernel(const float4* __restrict__ agg4, const float* __restrict__ inn,
                                unsigned short* __restrict__ abf) {
    long long tid = (long long)blockIdx.x * blockDim.x + threadIdx.x;  // N*32
    if (tid >= (long long)N_NODES * 32) return;
    int node = (int)(tid >> 5);
    int c = (int)(tid & 31);
    float s = inn[node];
    float4 v = agg4[tid];
    ushort4 o;
    o.x = f32_to_bf16(v.x * s);
    o.y = f32_to_bf16(v.y * s);
    o.z = f32_to_bf16(v.z * s);
    o.w = f32_to_bf16(v.w * s);
    *(ushort4*)(abf + (size_t)node * D + c * 4) = o;
}

// ---------------- WMMA GEMM: Y = relu(Abf @ Wt^T + bias), 16x128 tile per block ----------
__global__ __launch_bounds__(256) void wmma_matmul_kernel(
    const unsigned short* __restrict__ Abf,   // [M][128] bf16 row-major
    const unsigned short* __restrict__ Wt,    // [128 n][128 k] bf16 (transposed weights)
    const float* __restrict__ bias,           // [128]
    float* __restrict__ Y)                    // [M][128] f32
{
    const int wave = threadIdx.x >> 5;        // 8 waves -> 8 N-tiles of 16
    const int lane = threadIdx.x & 31;
    const int m0 = blockIdx.x << 4;
    const int n0 = wave << 4;
    const int lm  = lane & 15;
    const int lhi = lane >> 4;                // half-wave select per ISA fragment layout

    const unsigned short* arow = Abf + (size_t)(m0 + lm) * D;         // A: row = M
    const unsigned short* brow = Wt + (size_t)(n0 + lm) * D + lhi * 16; // B: row = N

    v8f acc = {};
#pragma unroll
    for (int kk = 0; kk < D; kk += 32) {
        BfFrag a, b;
        // A 16x32: lane holds M = m0+lm, 16 contiguous K at kk + 16*lhi
        a.u[0] = *(const uint4*)(arow + kk + lhi * 16);
        a.u[1] = *(const uint4*)(arow + kk + lhi * 16 + 8);
        // B 32x16: lane holds N = n0+lm, 16 contiguous K at kk + 16*lhi
        b.u[0] = *(const uint4*)(brow + kk);
        b.u[1] = *(const uint4*)(brow + kk + 8);
        acc = __builtin_amdgcn_wmma_f32_16x16x32_bf16(
            false, a.v, false, b.v, (short)0, acc, false, false);
    }
    const float bv = bias[n0 + lm];
#pragma unroll
    for (int r = 0; r < 8; ++r) {
        int mrow = m0 + r + lhi * 8;          // C/D: M = r + 8*(lane>=16), N = lane&15
        Y[(size_t)mrow * D + n0 + lm] = fmaxf(acc[r] + bv, 0.0f);
    }
}

// ---------------- per-graph mean pooling ----------------
__global__ void pool_kernel(const float4* __restrict__ h4, const int* __restrict__ n2g,
                            float* __restrict__ gsum, float* __restrict__ gcount) {
    long long tid = (long long)blockIdx.x * blockDim.x + threadIdx.x;  // N*32
    if (tid >= (long long)N_NODES * 32) return;
    int node = (int)(tid >> 5);
    int c = (int)(tid & 31);
    int g = n2g[node];
    float4 v = h4[tid];
    float* gs = gsum + (size_t)g * D + c * 4;
    atomicAdd(gs + 0, v.x);
    atomicAdd(gs + 1, v.y);
    atomicAdd(gs + 2, v.z);
    atomicAdd(gs + 3, v.w);
    if (c == 0) atomicAdd(&gcount[g], 1.0f);
}

__global__ void mean_kernel(const float* __restrict__ gsum, const float* __restrict__ gcount,
                            float* __restrict__ hg) {
    int tid = blockIdx.x * blockDim.x + threadIdx.x;  // G*128
    if (tid >= N_GRAPHS * D) return;
    int g = tid >> 7;
    hg[tid] = gsum[tid] / fmaxf(gcount[g], 1.0f);
}

// ---------------- tiny classifier matmuls (fp32 exact) ----------------
__global__ void mlp_kernel(const float* __restrict__ X, const float* __restrict__ W,
                           const float* __restrict__ b, float* __restrict__ Y,
                           int Nout, int dorelu) {
    int tid = blockIdx.x * blockDim.x + threadIdx.x;  // G * Nout
    if (tid >= N_GRAPHS * Nout) return;
    int g = tid / Nout, n = tid % Nout;
    float acc = b[n];
    const float* xr = X + (size_t)g * D;
    for (int k = 0; k < D; ++k) acc = fmaf(xr[k], W[k * Nout + n], acc);
    Y[tid] = dorelu ? fmaxf(acc, 0.0f) : acc;
}

extern "C" void kernel_launch(void* const* d_in, const int* in_sizes, int n_in,
                              void* d_out, int out_size, void* d_ws, size_t ws_size,
                              hipStream_t stream) {
    const float* h    = (const float*)d_in[0];
    const int*   src  = (const int*)  d_in[1];
    const int*   dst  = (const int*)  d_in[2];
    const int*   n2g  = (const int*)  d_in[3];
    const float* W1   = (const float*)d_in[4];
    const float* b1   = (const float*)d_in[5];
    const float* W2   = (const float*)d_in[6];
    const float* b2   = (const float*)d_in[7];
    const float* Wc1  = (const float*)d_in[8];
    const float* bc1  = (const float*)d_in[9];
    const float* Wc2  = (const float*)d_in[10];
    const float* bc2  = (const float*)d_in[11];
    const float* Wc3  = (const float*)d_in[12];
    const float* bc3  = (const float*)d_in[13];
    float* out = (float*)d_out;

    const size_t ND_BYTES = (size_t)N_NODES * D * sizeof(float);    // 51.2 MB
    char* w = (char*)d_ws;
    float*          bufA   = (float*)(w);                 // hn (fp32), later abf (bf16)
    unsigned short* abf    = (unsigned short*)(w);
    float*          bufB   = (float*)(w + ND_BYTES);      // agg, later layer output
    size_t off = 2 * ND_BYTES;
    float* outn  = (float*)(w + off); off += (size_t)N_NODES * 4;
    float* inn   = (float*)(w + off); off += (size_t)N_NODES * 4;
    unsigned short* Wt1 = (unsigned short*)(w + off); off += (size_t)D * D * 2;
    unsigned short* Wt2 = (unsigned short*)(w + off); off += (size_t)D * D * 2;
    float* gsum   = (float*)(w + off); off += (size_t)N_GRAPHS * D * 4;
    float* gcount = (float*)(w + off); off += 256;
    float* hg     = (float*)(w + off); off += (size_t)N_GRAPHS * D * 4;
    float* x1     = (float*)(w + off); off += (size_t)N_GRAPHS * D * 4;
    float* x2     = (float*)(w + off); off += (size_t)N_GRAPHS * D * 4;

    const int B = 256;
    const int gridE   = (N_EDGES + B - 1) / B;
    const int gridN   = (N_NODES + B - 1) / B;
    const int gridN32 = (int)(((long long)N_NODES * 32 + B - 1) / B);
    const int gridE32 = (int)(((long long)N_EDGES * 32 + B - 1) / B);
    const int gridMM  = N_NODES / 16;   // 6250 tiles of 16 rows, exact

    // degrees -> norms (recomputed every call; deterministic)
    hipMemsetAsync(outn, 0, (size_t)N_NODES * 8, stream);  // outn + inn contiguous
    deg_kernel<<<gridE, B, 0, stream>>>(src, dst, outn, inn);
    norm_kernel<<<gridN, B, 0, stream>>>(outn, inn);
    wconv_kernel<<<(2 * D * D + B - 1) / B, B, 0, stream>>>(W1, W2, Wt1, Wt2);

    // ---- layer 1 ----
    scale_kernel<<<gridN32, B, 0, stream>>>((const float4*)h, outn, (float4*)bufA);
    hipMemsetAsync(bufB, 0, ND_BYTES, stream);
    scatter_kernel<<<gridE32, B, 0, stream>>>((const float4*)bufA, src, dst, bufB);
    finalize_kernel<<<gridN32, B, 0, stream>>>((const float4*)bufB, inn, abf);
    wmma_matmul_kernel<<<gridMM, B, 0, stream>>>(abf, Wt1, b1, bufB);

    // ---- layer 2 ----
    scale_kernel<<<gridN32, B, 0, stream>>>((const float4*)bufB, outn, (float4*)bufA);
    hipMemsetAsync(bufB, 0, ND_BYTES, stream);
    scatter_kernel<<<gridE32, B, 0, stream>>>((const float4*)bufA, src, dst, bufB);
    finalize_kernel<<<gridN32, B, 0, stream>>>((const float4*)bufB, inn, abf);
    wmma_matmul_kernel<<<gridMM, B, 0, stream>>>(abf, Wt2, b2, bufB);

    // ---- mean pooling ----
    hipMemsetAsync(gsum, 0, (size_t)N_GRAPHS * D * 4 + 256, stream);  // gsum + gcount
    pool_kernel<<<gridN32, B, 0, stream>>>((const float4*)bufB, n2g, gsum, gcount);
    mean_kernel<<<(N_GRAPHS * D + B - 1) / B, B, 0, stream>>>(gsum, gcount, hg);

    // ---- classifier MLP (fp32) ----
    mlp_kernel<<<(N_GRAPHS * D + B - 1) / B, B, 0, stream>>>(hg, Wc1, bc1, x1, D, 1);
    mlp_kernel<<<(N_GRAPHS * D + B - 1) / B, B, 0, stream>>>(x1, Wc2, bc2, x2, D, 1);
    mlp_kernel<<<1, 64, 0, stream>>>(x2, Wc3, bc3, out, 1, 0);
}
